// EnergyAttention_15582141350436
// MI455X (gfx1250) — matrix operand order repaired and checked by reading, not compiled, not run on hardware
//
#include <hip/hip_runtime.h>

// ---------------------------------------------------------------------------
// Problem constants (from reference)
// ---------------------------------------------------------------------------
#define QN 2048
#define KN 4096
#define DD 1024
#define HH 16
#define HD 64
#define STEPS 5
#define STEP_SIZE 0.1f
#define BETA_SC 0.125f // 1/sqrt(64)

typedef unsigned short u16;
typedef __bf16 bf16_t;
typedef bf16_t v16bf __attribute__((ext_vector_type(16)));
typedef float v8f __attribute__((ext_vector_type(8)));
typedef unsigned int v4u __attribute__((ext_vector_type(4)));

union Frag {
  v16bf v;
  v4u q[2];
};

__device__ __forceinline__ u16 f2bf(float x) {
  unsigned int u = __float_as_uint(x);
  u += 0x7FFFu + ((u >> 16) & 1u); // round-to-nearest-even
  return (u16)(u >> 16);
}

__device__ __forceinline__ unsigned int pack2bf(float lo, float hi) {
  return (unsigned int)f2bf(lo) | ((unsigned int)f2bf(hi) << 16);
}

__device__ __forceinline__ v8f wmma_bf16(const Frag& a, const Frag& b, v8f c) {
  return __builtin_amdgcn_wmma_f32_16x16x32_bf16(
      /*neg_a=*/false, a.v, /*neg_b=*/false, b.v,
      /*c_mod=*/(short)0, c, /*reuse_a=*/false, /*reuse_b=*/false);
}

// ---------------------------------------------------------------------------
// f32 -> bf16 conversion
// ---------------------------------------------------------------------------
__global__ void cvt_f32_bf16(const float* __restrict__ in, u16* __restrict__ out, int n) {
  int i = blockIdx.x * 256 + threadIdx.x;
  if (i < n) out[i] = f2bf(in[i]);
}

// Qf32 [q][h*64+z] -> Qbf head-major [h][q][z], pre-scaled by beta
__global__ void snapshot_q(const float* __restrict__ Qf32, u16* __restrict__ Qbf) {
  int i = blockIdx.x * 256 + threadIdx.x;
  if (i >= QN * DD) return;
  int q = i >> 10, j = i & 1023;
  int h = j >> 6, z = j & 63;
  Qbf[((size_t)h * QN + q) * HD + z] = f2bf(Qf32[i] * BETA_SC);
}

// ---------------------------------------------------------------------------
// Generic bf16 GEMM: C[m][n] = sum_k A[m][k] * B[n][k],  Kdim = 1024
// MODE 0: store f32 row-major C [M][1024]
// MODE 1: K projection epilogue -> Kh [16][M][64] bf16 and Kt [16][64][M] bf16
// Block: 256 threads (8 waves), block tile 128x64, wave tile 32x32 (2x2 WMMA)
// ---------------------------------------------------------------------------
template <int MODE>
__global__ __launch_bounds__(256) void gemm_bf16_nt(
    const u16* __restrict__ A, const u16* __restrict__ B,
    float* __restrict__ Cf, u16* __restrict__ Kh, u16* __restrict__ Kt, int M) {
  __shared__ u16 As[128 * 32]; // 8 KB
  __shared__ u16 Bs[64 * 32];  // 4 KB

  const int tid = threadIdx.x;
  const int wave = tid >> 5, lane = tid & 31;
  const int half = lane >> 4, lrow = lane & 15;
  const int bm = blockIdx.x * 128;
  const int bn = blockIdx.y * 64;
  const int wm = (wave >> 1) * 32; // 4 waves along M
  const int wn = (wave & 1) * 32;  // 2 waves along N

  v8f c[2][2] = {};

  for (int k0 = 0; k0 < DD; k0 += 32) {
#pragma unroll
    for (int i = tid; i < 512; i += 256) {
      int r = i >> 2, cb = (i & 3) * 8;
      *(v4u*)&As[r * 32 + cb] = *(const v4u*)&A[(size_t)(bm + r) * DD + k0 + cb];
    }
    {
      int r = tid >> 2, cb = (tid & 3) * 8;
      *(v4u*)&Bs[r * 32 + cb] = *(const v4u*)&B[(size_t)(bn + r) * DD + k0 + cb];
    }
    __syncthreads();

    Frag af[2];
#pragma unroll
    for (int i = 0; i < 2; ++i) {
      const u16* p = &As[(wm + i * 16 + lrow) * 32 + half * 8];
      af[i].q[0] = *(const v4u*)p;
      af[i].q[1] = *(const v4u*)(p + 16);
    }
    Frag bf[2];
#pragma unroll
    for (int j = 0; j < 2; ++j) {
      const u16* p = &Bs[(wn + j * 16 + lrow) * 32 + half * 8];
      bf[j].q[0] = *(const v4u*)p;
      bf[j].q[1] = *(const v4u*)(p + 16);
    }
#pragma unroll
    for (int i = 0; i < 2; ++i)
#pragma unroll
      for (int j = 0; j < 2; ++j) c[i][j] = wmma_bf16(af[i], bf[j], c[i][j]);
    __syncthreads();
  }

#pragma unroll
  for (int i = 0; i < 2; ++i)
#pragma unroll
    for (int j = 0; j < 2; ++j)
#pragma unroll
      for (int r = 0; r < 8; ++r) {
        int m = bm + wm + i * 16 + half * 8 + r;
        int n = bn + wn + j * 16 + lrow;
        float v = c[i][j][r];
        if (MODE == 0) {
          Cf[(size_t)m * DD + n] = v;
        } else {
          u16 bv = f2bf(v);
          int h = n >> 6, z = n & 63;
          Kh[((size_t)h * M + m) * HD + z] = bv;
          Kt[((size_t)h * HD + z) * M + m] = bv;
        }
      }
}

// ---------------------------------------------------------------------------
// One energy step (transposed-score formulation, 2 query-subtiles per wave):
//   S^T = K Q^T   (rows = keys -> key dim lives in-lane: cheap softmax)
//   acc^T = K^T P^T  (rows = z)
//   Q += STEP * acc^T / l
// Grid: 256 blocks (16 q-blocks x 16 heads), 128 threads (4 waves).
// One wave owns 32 queries (2 subtiles); keys streamed in chunks of 64.
// Every K fragment (Kh and Kt) is reused by both subtiles:
// per chunk: 32 WMMA vs 32 global 16B loads, 4 shuffles, 8+8 DS ops.
// ---------------------------------------------------------------------------
__global__ __launch_bounds__(128) void attn_step(
    const u16* __restrict__ Qbf, // [16][QN][64]  (pre-scaled by beta)
    const u16* __restrict__ Kh,  // [16][KN][64]
    const u16* __restrict__ Kt,  // [16][64][KN]
    float* __restrict__ Qf32) {  // [QN][1024]
  __shared__ u16 pT[4][2][16 * 64]; // per-wave/subtile P^T: [q][key], 16 KB

  const int wave = threadIdx.x >> 5, lane = threadIdx.x & 31;
  const int half = lane >> 4, lrow = lane & 15;
  const int h = blockIdx.x & 15;
  const int qbase = (blockIdx.x >> 4) * 128 + wave * 32;

  const u16* kb = Kh + (size_t)h * KN * HD;
  const u16* ktb = Kt + (size_t)h * HD * KN;

  // Q as B-operand of S^T: B[z, q] -> lane = q col, z = half*8 .. +7 and 16 + half*8 .. +7
  Frag qb[2][2];
#pragma unroll
  for (int u = 0; u < 2; ++u)
#pragma unroll
    for (int f = 0; f < 2; ++f) {
      const u16* p =
          Qbf + ((size_t)h * QN + qbase + u * 16 + lrow) * HD + f * 32 + half * 8;
      qb[u][f].q[0] = *(const v4u*)p;
      qb[u][f].q[1] = *(const v4u*)(p + 16);
    }

  v8f acc[2][4] = {}; // acc^T per subtile: 4 z-tiles of 16x16, cols = q
  float mrun[2] = {-1e30f, -1e30f};
  float lrun[2] = {0.0f, 0.0f};

  for (int k0 = 0; k0 < KN; k0 += 64) {
    // ---- S^T: 4 key-tiles x 2 z-fragments, K fragment shared by subtiles --
    v8f s[2][4] = {};
#pragma unroll
    for (int t = 0; t < 4; ++t)
#pragma unroll
      for (int f = 0; f < 2; ++f) {
        Frag ak; // A = K tile: lane = key row, z contiguous per A-layout
        const u16* p = kb + (size_t)(k0 + t * 16 + lrow) * HD + f * 32 + half * 8;
        ak.q[0] = *(const v4u*)p;
        ak.q[1] = *(const v4u*)(p + 16);
#pragma unroll
        for (int u = 0; u < 2; ++u) s[u][t] = wmma_bf16(ak, qb[u][f], s[u][t]);
      }

    // ---- per-lane online softmax (32 keys in-lane + 1 exchange) -----------
#pragma unroll
    for (int u = 0; u < 2; ++u) {
      float mx = -1e30f;
#pragma unroll
      for (int t = 0; t < 4; ++t)
#pragma unroll
        for (int r = 0; r < 8; ++r) mx = fmaxf(mx, s[u][t][r]);
      mx = fmaxf(mx, __shfl_xor(mx, 16, 32));
      float mn = fmaxf(mrun[u], mx);
      if (__any(mx > mrun[u])) { // wave-uniform: rescale only when max moves
        float corr = __expf(mrun[u] - mn);
        lrun[u] *= corr;
#pragma unroll
        for (int t = 0; t < 4; ++t)
#pragma unroll
          for (int r = 0; r < 8; ++r) acc[u][t][r] *= corr;
      }
      mrun[u] = mn;

      float rs = 0.0f;
#pragma unroll
      for (int t = 0; t < 4; ++t) {
        float p8[8];
#pragma unroll
        for (int r = 0; r < 8; ++r) {
          p8[r] = __expf(s[u][t][r] - mn);
          rs += p8[r];
        }
        // packed store: 8 bf16 = 16 B -> pT[q][key], keys t*16 + half*8 .. +7
        v4u pk;
        pk.x = pack2bf(p8[0], p8[1]);
        pk.y = pack2bf(p8[2], p8[3]);
        pk.z = pack2bf(p8[4], p8[5]);
        pk.w = pack2bf(p8[6], p8[7]);
        *(v4u*)&pT[wave][u][lrow * 64 + t * 16 + half * 8] = pk;
      }
      rs += __shfl_xor(rs, 16, 32);
      lrun[u] += rs;
    }
    __builtin_amdgcn_wave_barrier(); // keep DS store->load order (in-order DS)

    // ---- acc^T += Kt-tile (16z x 32kk) * P^T (32kk x 16q) -----------------
#pragma unroll
    for (int g = 0; g < 2; ++g) {
      Frag pb[2]; // B = P^T: lane = q col, kk = g*32 + half*8 .. +7 and g*32 + 16 + half*8 .. +7
#pragma unroll
      for (int u = 0; u < 2; ++u) {
        const u16* pp = &pT[wave][u][lrow * 64 + g * 32 + half * 8];
        pb[u].q[0] = *(const v4u*)pp;
        pb[u].q[1] = *(const v4u*)(pp + 16);
      }
#pragma unroll
      for (int t = 0; t < 4; ++t) {
        Frag ak; // A = Kt tile: lane = z row, kk contiguous (shared by subtiles)
        const u16* p = ktb + (size_t)(t * 16 + lrow) * KN + k0 + g * 32 + half * 8;
        ak.q[0] = *(const v4u*)p;
        ak.q[1] = *(const v4u*)(p + 16);
#pragma unroll
        for (int u = 0; u < 2; ++u) acc[u][t] = wmma_bf16(ak, pb[u], acc[u][t]);
      }
    }
  }

  // ---- q_new = q_old + STEP * acc^T / l -----------------------------------
#pragma unroll
  for (int u = 0; u < 2; ++u) {
    const float inv = STEP_SIZE / lrun[u];
#pragma unroll
    for (int t = 0; t < 4; ++t)
#pragma unroll
      for (int r = 0; r < 8; ++r) {
        int z = t * 16 + r + half * 8;
        Qf32[(size_t)(qbase + u * 16 + lrow) * DD + h * HD + z] += acc[u][t][r] * inv;
      }
  }
}

// ---------------------------------------------------------------------------
// Host-side orchestration
// ---------------------------------------------------------------------------
extern "C" void kernel_launch(void* const* d_in, const int* in_sizes, int n_in,
                              void* d_out, int out_size, void* d_ws, size_t ws_size,
                              hipStream_t stream) {
  (void)in_sizes; (void)n_in; (void)out_size; (void)ws_size;
  const float* ctx = (const float*)d_in[0]; // [4096][1024]
  const float* tgt = (const float*)d_in[1]; // [2048][1024]
  const float* Wq = (const float*)d_in[2];  // [16][64][1024]
  const float* Wk = (const float*)d_in[3];  // [16][64][1024]
  const float* Wo = (const float*)d_in[4];  // [1024][1024]
  float* out = (float*)d_out;               // [2048][1024]

  char* w = (char*)d_ws;
  u16* ctx_bf = (u16*)w;   w += (size_t)KN * DD * 2; // 8 MB
  u16* tgt_bf = (u16*)w;   w += (size_t)QN * DD * 2; // 4 MB
  u16* Wq_bf = (u16*)w;    w += (size_t)DD * DD * 2; // 2 MB
  u16* Wk_bf = (u16*)w;    w += (size_t)DD * DD * 2; // 2 MB
  u16* Wo_bf = (u16*)w;    w += (size_t)DD * DD * 2; // 2 MB
  u16* Khh = (u16*)w;      w += (size_t)HH * KN * HD * 2; // 8 MB  [h][k][z]
  u16* Ktt = (u16*)w;      w += (size_t)HH * HD * KN * 2; // 8 MB  [h][z][k]
  float* Qf32 = (float*)w; w += (size_t)QN * DD * 4;      // 8 MB  == Q_flat
  u16* Qbf = (u16*)w;      w += (size_t)HH * QN * HD * 2; // 4 MB  [h][q][z]
  u16* Qflat_bf = (u16*)w; w += (size_t)QN * DD * 2;      // 4 MB

  // 1. fp32 -> bf16 conversions
  cvt_f32_bf16<<<(KN * DD + 255) / 256, 256, 0, stream>>>(ctx, ctx_bf, KN * DD);
  cvt_f32_bf16<<<(QN * DD + 255) / 256, 256, 0, stream>>>(tgt, tgt_bf, QN * DD);
  cvt_f32_bf16<<<(DD * DD + 255) / 256, 256, 0, stream>>>(Wq, Wq_bf, DD * DD);
  cvt_f32_bf16<<<(DD * DD + 255) / 256, 256, 0, stream>>>(Wk, Wk_bf, DD * DD);
  cvt_f32_bf16<<<(DD * DD + 255) / 256, 256, 0, stream>>>(Wo, Wo_bf, DD * DD);

  // 2. K projection -> dual layouts;  3. Q projection -> f32 master (= Q_flat)
  gemm_bf16_nt<1><<<dim3(KN / 128, DD / 64), 256, 0, stream>>>(ctx_bf, Wk_bf, nullptr, Khh, Ktt, KN);
  gemm_bf16_nt<0><<<dim3(QN / 128, DD / 64), 256, 0, stream>>>(tgt_bf, Wq_bf, Qf32, nullptr, nullptr, QN);

  // 4. five energy descent steps
  for (int s = 0; s < STEPS; ++s) {
    snapshot_q<<<(QN * DD + 255) / 256, 256, 0, stream>>>(Qf32, Qbf);
    attn_step<<<dim3((QN / 128) * HH), 128, 0, stream>>>(Qbf, Khh, Ktt, Qf32);
  }

  // 5. output projection: out = Q_flat @ Wo^T
  cvt_f32_bf16<<<(QN * DD + 255) / 256, 256, 0, stream>>>(Qf32, Qflat_bf, QN * DD);
  gemm_bf16_nt<0><<<dim3(QN / 128, DD / 64), 256, 0, stream>>>(Qflat_bf, Wo_bf, out, nullptr, nullptr, QN);
}
